// attention_encoding_network_87222195847684
// MI455X (gfx1250) — matrix operand order repaired and checked by reading, not compiled, not run on hardware
//
#include <hip/hip_runtime.h>
#include <cstdint>

typedef __bf16 bf16;
typedef __attribute__((ext_vector_type(16))) __bf16 v16bf;
typedef __attribute__((ext_vector_type(8)))  __bf16 v8bf;
typedef __attribute__((ext_vector_type(8)))  float  v8f;

#define LDS_PITCH 72   // padded pitch (elements) for 64-wide K tiles: 144B rows, 16B-aligned, conflict-free

// ---------------------------------------------------------------------------
// Elementwise / conversion kernels
// ---------------------------------------------------------------------------
__global__ void k_f32_to_bf16(const float* __restrict__ src, bf16* __restrict__ dst, int n) {
    int i = blockIdx.x * blockDim.x + threadIdx.x;
    if (i < n) dst[i] = (bf16)src[i];
}

// dst[c*rows + r] = src[r*cols + c]  (builds encode1^T in bf16)
__global__ void k_transpose_bf16(const float* __restrict__ src, bf16* __restrict__ dst,
                                 int rows, int cols) {
    int i = blockIdx.x * blockDim.x + threadIdx.x;
    if (i < rows * cols) {
        int r = i / cols, c = i % cols;
        dst[(size_t)c * rows + r] = (bf16)src[i];
    }
}

// prep(x): permute(0,2,3,1) -> (N,V,T*C) -> adaptive avg pool to IN_CH (window 16)
// x layout (N,C,T,V) = (4,64,64,1024); out (N,V,256) bf16
__global__ void k_prep(const float* __restrict__ x, bf16* __restrict__ out, int total) {
    int i = blockIdx.x * blockDim.x + threadIdx.x;
    if (i >= total) return;
    int ich = i & 255;
    int rest = i >> 8;
    int p = rest & 1023;
    int n = rest >> 10;
    float s = 0.f;
    int qbase = ich * 16;
#pragma unroll
    for (int w = 0; w < 16; ++w) {
        int flat = p * 4096 + qbase + w;       // index in (T,V,C) flat order
        int t   = flat >> 16;                  // / (V*C) = 65536
        int rem = flat & 65535;
        int v   = rem >> 6;                    // / C
        int c   = rem & 63;
        s += x[(((size_t)(n * 64 + c) * 64 + t) * 1024) + v];
    }
    out[i] = (bf16)(s * (1.0f / 16.0f));
}

// ---------------------------------------------------------------------------
// Generic WMMA bf16 GEMM:  C = scale * (A @ B[^T]) + bias, then LeakyReLU(slope)
// NN path: LDS-staged, double-buffered, K-tile 64 (2 WMMAs per barrier);
//          A tile via global_load_async_to_lds_b128 (ASYNCcnt),
//          B tile transposed into LDS (column-major, padded) for vector ds reads.
// NT path: register-direct (both operands contiguous in K).
// Block = 256 threads = 8 waves, block tile 32(M) x 64(N); wave tile 16x16.
// ---------------------------------------------------------------------------
template <bool TRANS_B>
__global__ __launch_bounds__(256)
void k_gemm(const bf16* __restrict__ A, const bf16* __restrict__ B,
            const float* __restrict__ bias,
            float* __restrict__ Cf, bf16* __restrict__ Cb,
            int M, int N, int K, int lda, int ldb, int ldc,
            long long sAo, long long sAi, long long sBo, long long sBi,
            long long sCo, long long sCi, int innerN,
            float scale, float slope) {
    int z  = blockIdx.z;
    int zo = z / innerN, zi = z % innerN;
    const bf16* Ab = A + zo * sAo + zi * sAi;
    const bf16* Bb = B + zo * sBo + zi * sBi;
    long long offC = (long long)zo * sCo + (long long)zi * sCi;

    int lane = threadIdx.x & 31;
    int wave = threadIdx.x >> 5;
    int wm = wave >> 2;                 // 0..1
    int wn = wave & 3;                  // 0..3
    int rowBase = blockIdx.y * 32 + wm * 16;
    int colBase = blockIdx.x * 64 + wn * 16;
    int mloc = lane & 15;
    int hi   = lane >> 4;

    v8f acc = {};

    if constexpr (TRANS_B) {
        // operand[k, ncol] = B[ncol, k]: contiguous in k for both A and B
        const bf16* Arow = Ab + (size_t)(rowBase + mloc) * lda + hi * 8;
        const bf16* Brow = Bb + (size_t)(colBase + mloc) * ldb + hi * 8;
        for (int k0 = 0; k0 < K; k0 += 32) {
            v8bf a0 = *(const v8bf*)(Arow + k0);
            v8bf a1 = *(const v8bf*)(Arow + k0 + 16);
            v8bf b0 = *(const v8bf*)(Brow + k0);
            v8bf b1 = *(const v8bf*)(Brow + k0 + 16);
            v16bf af, bfr;
#pragma unroll
            for (int j = 0; j < 8; ++j) {
                af[j] = a0[j]; af[j + 8] = a1[j];
                bfr[j] = b0[j]; bfr[j + 8] = b1[j];
            }
            acc = __builtin_amdgcn_wmma_f32_16x16x32_bf16(false, af, false, bfr,
                                                          (short)0, acc, false, false);
        }
    } else {
        __shared__ bf16 ldsA[2][32 * LDS_PITCH];   // A tile 32(M) x 64(K), row-major
        __shared__ bf16 ldsB[2][64 * LDS_PITCH];   // B tile 64(N) x 64(K), column-major (transposed)
        int tid = threadIdx.x;
        // A staging: 256 threads x 1 async b128 chunk: row = tid>>3, kc = (tid&7)*8
        int arow = tid >> 3;
        int akc  = (tid & 7) * 8;
        const bf16* gA = Ab + (size_t)(blockIdx.y * 32 + arow) * lda + akc;
        // B staging: 2 chunks per thread (512 chunks of 8 elements)
        const bf16* gB0 = Bb + (size_t)blockIdx.x * 64;

        auto stage = [&](int k0, int buf) {
            {
                unsigned ldsoff =
                    (unsigned)(size_t)(void*)&ldsA[buf][arow * LDS_PITCH + akc];
                unsigned long long ga = (unsigned long long)(const void*)(gA + k0);
                asm volatile("global_load_async_to_lds_b128 %0, %1, off"
                             :: "v"(ldsoff), "v"(ga) : "memory");
            }
#pragma unroll
            for (int h = 0; h < 2; ++h) {
                int c  = tid + h * 256;
                int kr = c >> 3;              // 0..63 (K row within tile)
                int nc = (c & 7) * 8;         // 0..56 (N within tile)
                v8bf tmp = *(const v8bf*)(gB0 + (size_t)(k0 + kr) * ldb + nc);
#pragma unroll
                for (int j = 0; j < 8; ++j)
                    ldsB[buf][(nc + j) * LDS_PITCH + kr] = tmp[j];
            }
            if (k0 + 128 < K)
                __builtin_prefetch(gB0 + (size_t)(k0 + 128 + (tid >> 3)) * ldb, 0, 3);
        };

        stage(0, 0);
        int nk = K >> 6;
        for (int i = 0; i < nk; ++i) {
            asm volatile("s_wait_asynccnt 0x0" ::: "memory");
            __syncthreads();             // publish tile i; separates reads(i-1) from writes(i+1)
            int cur = i & 1;
            if (i + 1 < nk) stage((i + 1) << 6, cur ^ 1);
            const bf16* apb = &ldsA[cur][(wm * 16 + mloc) * LDS_PITCH + hi * 8];
            const bf16* bpb = &ldsB[cur][(wn * 16 + mloc) * LDS_PITCH + hi * 8];
#pragma unroll
            for (int sub = 0; sub < 2; ++sub) {
                const bf16* ap = apb + sub * 32;
                const bf16* bp = bpb + sub * 32;
                v8bf a0 = *(const v8bf*)ap;
                v8bf a1 = *(const v8bf*)(ap + 16);
                v8bf b0 = *(const v8bf*)bp;
                v8bf b1 = *(const v8bf*)(bp + 16);
                v16bf af, bfr;
#pragma unroll
                for (int j = 0; j < 8; ++j) {
                    af[j] = a0[j]; af[j + 8] = a1[j];
                    bfr[j] = b0[j]; bfr[j + 8] = b1[j];
                }
                acc = __builtin_amdgcn_wmma_f32_16x16x32_bf16(false, af, false, bfr,
                                                              (short)0, acc, false, false);
            }
        }
    }

    // C/D layout: VGPR r -> row = rowBase + hi*8 + r, col = colBase + (lane&15)
    int col = colBase + mloc;
    float bv = bias ? bias[col] : 0.0f;
#pragma unroll
    for (int r = 0; r < 8; ++r) {
        int row = rowBase + hi * 8 + r;
        float v = acc[r] * scale + bv;
        v = (v >= 0.f) ? v : slope * v;
        size_t idx = (size_t)offC + (size_t)row * ldc + col;
        if (Cf) Cf[idx] = v;
        if (Cb) Cb[idx] = (bf16)v;
    }
}

// ---------------------------------------------------------------------------
// BatchNorm over (rows x cols): biased variance, per-column stats
// ---------------------------------------------------------------------------
__global__ void k_bn_stats(const float* __restrict__ h, float* __restrict__ mu,
                           float* __restrict__ rsig, int rows, int cols) {
    int j = blockIdx.x;
    float s = 0.f, s2 = 0.f;
    for (int r = threadIdx.x; r < rows; r += blockDim.x) {
        float v = h[(size_t)r * cols + j];
        s += v; s2 += v * v;
    }
    __shared__ float red[256], red2[256];
    red[threadIdx.x] = s; red2[threadIdx.x] = s2; __syncthreads();
    for (int o = 128; o > 0; o >>= 1) {
        if (threadIdx.x < o) { red[threadIdx.x] += red[threadIdx.x + o];
                               red2[threadIdx.x] += red2[threadIdx.x + o]; }
        __syncthreads();
    }
    if (threadIdx.x == 0) {
        float m = red[0] / rows;
        float var = red2[0] / rows - m * m;
        mu[j] = m;
        rsig[j] = rsqrtf(var + 1e-5f);
    }
}

__global__ void k_bn_apply(const float* __restrict__ h, const float* __restrict__ mu,
                           const float* __restrict__ rsig, const float* __restrict__ g,
                           const float* __restrict__ b, bf16* __restrict__ out,
                           int total, int cols) {
    int i = blockIdx.x * blockDim.x + threadIdx.x;
    if (i >= total) return;
    int j = i % cols;
    float v = (h[i] - mu[j]) * rsig[j] * g[j] + b[j];
    v = (v >= 0.f) ? v : 0.01f * v;
    out[i] = (bf16)v;
}

// ---------------------------------------------------------------------------
// Softmax: one block (256 thr) per row
// ---------------------------------------------------------------------------
__global__ void k_softmax_bf16(const float* __restrict__ S, bf16* __restrict__ P, int rowlen) {
    size_t base = (size_t)blockIdx.x * rowlen;
    __shared__ float red[256];
    float m = -3.0e38f;
    for (int j = threadIdx.x; j < rowlen; j += 256) m = fmaxf(m, S[base + j]);
    red[threadIdx.x] = m; __syncthreads();
    for (int o = 128; o > 0; o >>= 1) {
        if (threadIdx.x < o) red[threadIdx.x] = fmaxf(red[threadIdx.x], red[threadIdx.x + o]);
        __syncthreads();
    }
    m = red[0]; __syncthreads();
    float s = 0.f;
    for (int j = threadIdx.x; j < rowlen; j += 256) s += expf(S[base + j] - m);
    red[threadIdx.x] = s; __syncthreads();
    for (int o = 128; o > 0; o >>= 1) {
        if (threadIdx.x < o) red[threadIdx.x] += red[threadIdx.x + o];
        __syncthreads();
    }
    float inv = 1.0f / red[0];
    for (int j = threadIdx.x; j < rowlen; j += 256)
        P[base + j] = (bf16)(expf(S[base + j] - m) * inv);
}

__global__ void k_softmax_f32(float* __restrict__ S, int rowlen) {
    size_t base = (size_t)blockIdx.x * rowlen;
    __shared__ float red[256];
    float m = -3.0e38f;
    for (int j = threadIdx.x; j < rowlen; j += 256) m = fmaxf(m, S[base + j]);
    red[threadIdx.x] = m; __syncthreads();
    for (int o = 128; o > 0; o >>= 1) {
        if (threadIdx.x < o) red[threadIdx.x] = fmaxf(red[threadIdx.x], red[threadIdx.x + o]);
        __syncthreads();
    }
    m = red[0]; __syncthreads();
    float s = 0.f;
    for (int j = threadIdx.x; j < rowlen; j += 256) {
        float e = expf(S[base + j] - m);
        S[base + j] = e; s += e;
    }
    red[threadIdx.x] = s; __syncthreads();
    for (int o = 128; o > 0; o >>= 1) {
        if (threadIdx.x < o) red[threadIdx.x] += red[threadIdx.x + o];
        __syncthreads();
    }
    float inv = 1.0f / red[0];
    for (int j = threadIdx.x; j < rowlen; j += 256) S[base + j] *= inv;
}

// c2 = [a2; a1] i.e. swap halves of the row dimension of c1 (N,2048,512)
__global__ void k_halfswap(const bf16* __restrict__ c1, bf16* __restrict__ c2, int total) {
    int i = blockIdx.x * blockDim.x + threadIdx.x;
    if (i >= total) return;
    int c = i & 511;
    int row = (i >> 9) & 2047;
    int n = i >> 20;               // 2048*512 = 1<<20
    c2[i] = c1[((size_t)n << 20) + (((row + 1024) & 2047) << 9) + c];
}

// column sums of A (N, rows, cols): one block per (n, j)
__global__ void k_colsum(const float* __restrict__ A, float* __restrict__ colsum,
                         int rows, int cols) {
    int j = blockIdx.x % cols;
    int n = blockIdx.x / cols;
    const float* An = A + (size_t)n * rows * cols;
    float s = 0.f;
    for (int r = threadIdx.x; r < rows; r += 256) s += An[(size_t)r * cols + j];
    __shared__ float red[256];
    red[threadIdx.x] = s; __syncthreads();
    for (int o = 128; o > 0; o >>= 1) {
        if (threadIdx.x < o) red[threadIdx.x] += red[threadIdx.x + o];
        __syncthreads();
    }
    if (threadIdx.x == 0) colsum[blockIdx.x] = red[0];
}

__global__ void k_colscale(float* __restrict__ A, const float* __restrict__ colsum,
                           int rows, int cols, int total) {
    int i = blockIdx.x * blockDim.x + threadIdx.x;
    if (i >= total) return;
    int j = i % cols;
    int n = i / (rows * cols);
    float c = colsum[n * cols + j];
    A[i] *= (c > 0.f) ? (1.0f / c) : 0.0f;
}

// ---------------------------------------------------------------------------
// Host orchestration
// ---------------------------------------------------------------------------
extern "C" void kernel_launch(void* const* d_in, const int* in_sizes, int n_in,
                              void* d_out, int out_size, void* d_ws, size_t ws_size,
                              hipStream_t stream) {
    (void)in_sizes; (void)n_in; (void)out_size; (void)ws_size;
    const int Nb = 4, Vv = 1024, Ee = 4096, INC = 256, OUT = 512, Hh = 8, Ll = 1024;

    const float* x1      = (const float*)d_in[0];
    const float* x2      = (const float*)d_in[1];
    const float* encode1 = (const float*)d_in[2];
    const float* encode2 = (const float*)d_in[3];

    // ---- workspace carve (bump allocator, 256B aligned) ----
    char* wp = (char*)d_ws;
    auto alloc = [&](size_t bytes) -> void* {
        void* r = wp; wp += (bytes + 255) & ~(size_t)255; return r;
    };
    bf16* e1b   = (bf16*)alloc((size_t)Ee * Vv * 2);
    bf16* e2b   = (bf16*)alloc((size_t)Ee * Vv * 2);
    bf16* e1Tb  = (bf16*)alloc((size_t)Ee * Vv * 2);
    bf16* x1pb  = (bf16*)alloc((size_t)Nb * Vv * INC * 2);
    bf16* x2pb  = (bf16*)alloc((size_t)Nb * Vv * INC * 2);
    bf16* wb[12];
    for (int i = 0; i < 12; ++i) wb[i] = (bf16*)alloc((size_t)OUT * OUT * 2);
    bf16* nodeA = (bf16*)alloc((size_t)Nb * Ee * OUT * 2);
    bf16* nodeB = (bf16*)alloc((size_t)Nb * Ee * OUT * 2);
    bf16* lin1b = (bf16*)alloc((size_t)Nb * Ee * OUT * 2);
    float* lin2f = (float*)alloc((size_t)Nb * Ee * OUT * 4);
    float* mu    = (float*)alloc(OUT * 4);
    float* rsig  = (float*)alloc(OUT * 4);
    bf16* hbnb  = (bf16*)alloc((size_t)Nb * Ee * OUT * 2);
    bf16* edge1 = (bf16*)alloc((size_t)Nb * Vv * OUT * 2);
    bf16* edge2 = (bf16*)alloc((size_t)Nb * Vv * OUT * 2);
    bf16* Qb    = (bf16*)alloc((size_t)Nb * Ll * OUT * 2);
    bf16* Kb    = (bf16*)alloc((size_t)Nb * Ll * OUT * 2);
    bf16* Vb    = (bf16*)alloc((size_t)Nb * Ll * OUT * 2);
    float* Sf   = (float*)alloc((size_t)Hh * Ll * Ll * 4);   // per-n score buffer
    bf16* Pb    = (bf16*)alloc((size_t)Hh * Ll * Ll * 2);
    bf16* xattb = (bf16*)alloc((size_t)Nb * Ll * OUT * 2);
    bf16* c1b   = (bf16*)alloc((size_t)Nb * 2048 * OUT * 2);
    bf16* c2b   = (bf16*)alloc((size_t)Nb * 2048 * OUT * 2);
    float* colsum = (float*)alloc((size_t)Nb * 2048 * 4);

    auto conv = [&](const float* s, bf16* d, int n) {
        k_f32_to_bf16<<<(n + 255) / 256, 256, 0, stream>>>(s, d, n);
    };
    auto gemm = [&](bool nt, const bf16* A, const bf16* B, const float* bias,
                    float* Cf, bf16* Cb, int M, int Nc, int K,
                    int lda, int ldb, int ldc,
                    long long sAo, long long sAi, long long sBo, long long sBi,
                    long long sCo, long long sCi, int inner, int batches,
                    float scale, float slope) {
        dim3 g(Nc / 64, M / 32, batches), blk(256);
        if (nt)
            k_gemm<true><<<g, blk, 0, stream>>>(A, B, bias, Cf, Cb, M, Nc, K,
                lda, ldb, ldc, sAo, sAi, sBo, sBi, sCo, sCi, inner, scale, slope);
        else
            k_gemm<false><<<g, blk, 0, stream>>>(A, B, bias, Cf, Cb, M, Nc, K,
                lda, ldb, ldc, sAo, sAi, sBo, sBi, sCo, sCi, inner, scale, slope);
    };

    // ---- conversions ----
    conv(encode1, e1b, Ee * Vv);
    conv(encode2, e2b, Ee * Vv);
    k_transpose_bf16<<<(Ee * Vv + 255) / 256, 256, 0, stream>>>(encode1, e1Tb, Ee, Vv);
    conv((const float*)d_in[4],  wb[0], OUT * OUT);   // mlp_w1
    conv((const float*)d_in[6],  wb[1], OUT * OUT);   // mlp_w2
    conv((const float*)d_in[10], wb[2], OUT * OUT);   // nlp_w1
    conv((const float*)d_in[12], wb[3], OUT * OUT);   // nlp_w2
    for (int a = 0; a < 2; ++a)
        for (int j = 0; j < 4; ++j)
            conv((const float*)d_in[16 + a * 8 + j * 2], wb[4 + a * 4 + j], OUT * OUT);

    // ---- prep / pooling ----
    {
        int tot = Nb * Vv * INC;
        k_prep<<<(tot + 255) / 256, 256, 0, stream>>>(x1, x1pb, tot);
        k_prep<<<(tot + 255) / 256, 256, 0, stream>>>(x2, x2pb, tot);
    }

    // ---- node encoding: node[n,e, 0:256]=encode1@xp ; [256:512]=encode2@xp ----
    const long long sB_x = (long long)Vv * INC, sC_n = (long long)Ee * OUT;
    gemm(false, e1b, x1pb, nullptr, nullptr, nodeA,       Ee, INC, Vv, Vv, INC, OUT,
         0, 0, sB_x, 0, sC_n, 0, 1, Nb, 1.f, 1.f);
    gemm(false, e2b, x1pb, nullptr, nullptr, nodeA + 256, Ee, INC, Vv, Vv, INC, OUT,
         0, 0, sB_x, 0, sC_n, 0, 1, Nb, 1.f, 1.f);
    gemm(false, e1b, x2pb, nullptr, nullptr, nodeB,       Ee, INC, Vv, Vv, INC, OUT,
         0, 0, sB_x, 0, sC_n, 0, 1, Nb, 1.f, 1.f);
    gemm(false, e2b, x2pb, nullptr, nullptr, nodeB + 256, Ee, INC, Vv, Vv, INC, OUT,
         0, 0, sB_x, 0, sC_n, 0, 1, Nb, 1.f, 1.f);

    // ---- MLP branches (mlp on nodeA -> edge1, nlp on nodeB -> edge2) ----
    for (int b = 0; b < 2; ++b) {
        const bf16* node = b ? nodeB : nodeA;
        bf16* edge = b ? edge2 : edge1;
        int base = 4 + 6 * b;
        const float* b1 = (const float*)d_in[base + 1];
        const float* b2 = (const float*)d_in[base + 3];
        const float* ga = (const float*)d_in[base + 4];
        const float* be = (const float*)d_in[base + 5];
        gemm(false, node, wb[b * 2 + 0], b1, nullptr, lin1b, Ee, OUT, OUT,
             OUT, OUT, OUT, sC_n, 0, 0, 0, sC_n, 0, 1, Nb, 1.f, 0.01f);
        gemm(false, lin1b, wb[b * 2 + 1], b2, lin2f, nullptr, Ee, OUT, OUT,
             OUT, OUT, OUT, sC_n, 0, 0, 0, sC_n, 0, 1, Nb, 1.f, 1.f);
        k_bn_stats<<<OUT, 256, 0, stream>>>(lin2f, mu, rsig, Nb * Ee, OUT);
        {
            int tot = Nb * Ee * OUT;
            k_bn_apply<<<(tot + 255) / 256, 256, 0, stream>>>(lin2f, mu, rsig, ga, be,
                                                              hbnb, tot, OUT);
        }
        gemm(false, e1Tb, hbnb, nullptr, nullptr, edge, Vv, OUT, Ee,
             Ee, OUT, OUT, 0, 0, sC_n, 0, (long long)Vv * OUT, 0, 1, Nb,
             1.0f / (float)Vv, 1.f);
    }

    // ---- attention blocks ----
    const long long sL = (long long)Ll * OUT;   // 1024*512
    for (int a = 0; a < 2; ++a) {
        const bf16* qsrc = a ? edge1 : edge2;   // att1: q=e2 ; att2: q=e1
        const bf16* ksrc = a ? edge2 : edge1;   // att1: k,v=e1 ; att2: k,v=e2
        int base = 16 + a * 8;
        const float* bq = (const float*)d_in[base + 1];
        const float* bk = (const float*)d_in[base + 3];
        const float* bv = (const float*)d_in[base + 5];
        const float* bf_ = (const float*)d_in[base + 7];
        gemm(false, qsrc, wb[4 + a * 4 + 0], bq, nullptr, Qb, Ll, OUT, OUT,
             OUT, OUT, OUT, sL, 0, 0, 0, sL, 0, 1, Nb, 1.f, 1.f);
        gemm(false, ksrc, wb[4 + a * 4 + 1], bk, nullptr, Kb, Ll, OUT, OUT,
             OUT, OUT, OUT, sL, 0, 0, 0, sL, 0, 1, Nb, 1.f, 1.f);
        gemm(false, ksrc, wb[4 + a * 4 + 2], bv, nullptr, Vb, Ll, OUT, OUT,
             OUT, OUT, OUT, sL, 0, 0, 0, sL, 0, 1, Nb, 1.f, 1.f);
        for (int n = 0; n < Nb; ++n) {
            const bf16* Qn = Qb + (size_t)n * sL;
            const bf16* Kn = Kb + (size_t)n * sL;
            const bf16* Vn = Vb + (size_t)n * sL;
            // S[h] = (Q_h @ K_h^T) / 8 : M=N=1024, K=64
            gemm(true, Qn, Kn, nullptr, Sf, nullptr, Ll, Ll, 64,
                 OUT, OUT, Ll, 0, 64, 0, 64, 0, (long long)Ll * Ll, Hh, Hh,
                 0.125f, 1.f);
            k_softmax_bf16<<<Hh * Ll, 256, 0, stream>>>(Sf, Pb, Ll);
            // x_att[h] = P_h @ V_h : M=1024, N=64, K=1024
            gemm(false, Pb, Vn, nullptr, nullptr, xattb + (size_t)n * sL,
                 Ll, 64, Ll, Ll, OUT, OUT,
                 0, (long long)Ll * Ll, 0, 64, 0, 64, Hh, Hh, 1.f, 1.f);
        }
        bf16* cdst = c1b + (a ? (size_t)Ll * OUT : 0);
        gemm(false, xattb, wb[4 + a * 4 + 3], bf_, nullptr, cdst, Ll, OUT, OUT,
             OUT, OUT, OUT, sL, 0, 0, 0, (long long)2048 * OUT, 0, 1, Nb, 1.f, 1.f);
    }

    // ---- final: A = softmax(c1 @ c2^T), column-degree normalization ----
    {
        int tot = Nb * 2048 * OUT;
        k_halfswap<<<(tot + 255) / 256, 256, 0, stream>>>(c1b, c2b, tot);
    }
    float* outf = (float*)d_out;
    gemm(true, c1b, c2b, nullptr, outf, nullptr, 2048, 2048, OUT,
         OUT, OUT, 2048, (long long)2048 * OUT, 0, (long long)2048 * OUT, 0,
         (long long)2048 * 2048, 0, 1, Nb, 1.f, 1.f);
    k_softmax_f32<<<Nb * 2048, 256, 0, stream>>>(outf, 2048);
    k_colsum<<<Nb * 2048, 256, 0, stream>>>(outf, colsum, 2048, 2048);
    {
        int tot = Nb * 2048 * 2048;
        k_colscale<<<(tot + 255) / 256, 256, 0, stream>>>(outf, colsum, 2048, 2048, tot);
    }
}
